// NaiveManifoldConstrainedHyperConnection_76647986365059
// MI455X (gfx1250) — matrix operand order repaired and checked by reading, not compiled, not run on hardware
//
#include <hip/hip_runtime.h>
#include <stdint.h>

#define Wd   4
#define Dm   2048
#define NTOK 8192
#define NOUT 24

#define BM 128
#define BN 128
#define BK 32
#define LDSS 80   // padded LDS row stride in bytes (32 bf16 = 64B data + 16B pad)

typedef __attribute__((ext_vector_type(16))) __bf16 v16bf;
typedef __attribute__((ext_vector_type(8)))  float  v8f;

union Frag { uint4 u4[2]; v16bf v; };

__device__ __forceinline__ uint16_t f32_to_bf16(float f) {
    unsigned int u = __float_as_uint(f);
    u = (u + 0x7FFFu + ((u >> 16) & 1u)) >> 16;
    return (uint16_t)u;
}

__device__ __forceinline__ void softmax4(const float* v, float* o) {
    float m = fmaxf(fmaxf(v[0], v[1]), fmaxf(v[2], v[3]));
    float e0 = __expf(v[0] - m), e1 = __expf(v[1] - m);
    float e2 = __expf(v[2] - m), e3 = __expf(v[3] - m);
    float s = 1.0f / (e0 + e1 + e2 + e3);
    o[0] = e0 * s; o[1] = e1 * s; o[2] = e2 * s; o[3] = e3 * s;
}

// ---------------- Kernel 0: op_w fp32 -> bf16 ----------------
__global__ __launch_bounds__(256) void k_convert(const float* __restrict__ w,
                                                 uint16_t* __restrict__ o) {
    size_t i = ((size_t)blockIdx.x * 256 + threadIdx.x) * 8;
    float4 a = *(const float4*)(w + i);
    float4 b = *(const float4*)(w + i + 4);
    union { uint16_t s[8]; uint4 u; } p;
    p.s[0] = f32_to_bf16(a.x); p.s[1] = f32_to_bf16(a.y);
    p.s[2] = f32_to_bf16(a.z); p.s[3] = f32_to_bf16(a.w);
    p.s[4] = f32_to_bf16(b.x); p.s[5] = f32_to_bf16(b.y);
    p.s[6] = f32_to_bf16(b.z); p.s[7] = f32_to_bf16(b.w);
    *(uint4*)(o + i) = p.u;
}

// ---------------- Kernel 1: router + aggregate ----------------
// 256 threads = 8 waves; each wave handles 4 tokens; grid = NTOK/32 = 256
__global__ __launch_bounds__(256) void k_router(const float* __restrict__ x,
                                                const float* __restrict__ rw,
                                                const float* __restrict__ rb,
                                                const float* __restrict__ resb,
                                                uint16_t* __restrict__ agg,
                                                float* __restrict__ params) {
    int lane = threadIdx.x & 31;
    int wave = threadIdx.x >> 5;
    int tok0 = blockIdx.x * 32 + wave * 4;

    float acc[4][24];
#pragma unroll
    for (int tt = 0; tt < 4; tt++)
#pragma unroll
        for (int r = 0; r < 24; r++) acc[tt][r] = 0.0f;

    // phase 1: mean over W + 24 router dot products (lanes stride D)
    for (int c = 0; c < Dm / 32; c++) {
        int d = c * 32 + lane;
        float m[4];
#pragma unroll
        for (int tt = 0; tt < 4; tt++) {
            const float* xp = x + (size_t)(tok0 + tt) * Wd * Dm + d;
            m[tt] = 0.25f * (xp[0] + xp[Dm] + xp[2 * Dm] + xp[3 * Dm]);
        }
#pragma unroll
        for (int r = 0; r < 24; r++) {
            float wv = rw[r * Dm + d];
#pragma unroll
            for (int tt = 0; tt < 4; tt++) acc[tt][r] = fmaf(m[tt], wv, acc[tt][r]);
        }
    }

    // wave32 butterfly reduction; all lanes end with the full sums
#pragma unroll
    for (int tt = 0; tt < 4; tt++)
#pragma unroll
        for (int r = 0; r < 24; r++) {
            float v = acc[tt][r];
            v += __shfl_xor(v, 16, 32);
            v += __shfl_xor(v, 8, 32);
            v += __shfl_xor(v, 4, 32);
            v += __shfl_xor(v, 2, 32);
            v += __shfl_xor(v, 1, 32);
            acc[tt][r] = v + rb[r];
        }

#pragma unroll
    for (int tt = 0; tt < 4; tt++) {
        int token = tok0 + tt;
        float pre[4], post[4];
        softmax4(&acc[tt][0], pre);
        softmax4(&acc[tt][4], post);

        float rm[16];
#pragma unroll
        for (int i = 0; i < 16; i++) rm[i] = __expf(acc[tt][8 + i] + resb[i]);
#pragma unroll
        for (int it = 0; it < 4; it++) {
#pragma unroll
            for (int i = 0; i < 4; i++) {
                float s = rm[i * 4] + rm[i * 4 + 1] + rm[i * 4 + 2] + rm[i * 4 + 3];
                s = 1.0f / fmaxf(s, 1e-6f);
                rm[i * 4] *= s; rm[i * 4 + 1] *= s; rm[i * 4 + 2] *= s; rm[i * 4 + 3] *= s;
            }
#pragma unroll
            for (int j = 0; j < 4; j++) {
                float s = rm[j] + rm[4 + j] + rm[8 + j] + rm[12 + j];
                s = 1.0f / fmaxf(s, 1e-6f);
                rm[j] *= s; rm[4 + j] *= s; rm[8 + j] *= s; rm[12 + j] *= s;
            }
        }

        // phase 2: aggregated = sum_w pre[w] * x[w,:]  (bf16 for the WMMA GEMM)
        const float* xp = x + (size_t)token * Wd * Dm;
        for (int c = 0; c < Dm / 32; c++) {
            int d = c * 32 + lane;
            float ag = pre[0] * xp[d] + pre[1] * xp[Dm + d] +
                       pre[2] * xp[2 * Dm + d] + pre[3] * xp[3 * Dm + d];
            agg[(size_t)token * Dm + d] = f32_to_bf16(ag);
        }
        if (lane == 0) {
            float* pp = params + (size_t)token * 20;
#pragma unroll
            for (int i = 0; i < 4; i++) pp[i] = post[i];
#pragma unroll
            for (int i = 0; i < 16; i++) pp[4 + i] = rm[i];
        }
    }
}

// ---------------- Kernel 2: bf16 WMMA GEMM + fused epilogue ----------------
// grid = (N/BN = 16, M/BM = 64); 256 threads = 8 waves (2 M x 4 N), wave tile 64x32
// Register-staged double-buffered LDS pipeline: tile s+1 global loads are in
// flight while tile s is consumed by WMMAs; one barrier per K-step.
__global__ __launch_bounds__(256) void k_gemm(const uint16_t* __restrict__ agg,
                                              const uint16_t* __restrict__ wb,
                                              const float* __restrict__ opb,
                                              const float* __restrict__ params,
                                              const float* __restrict__ x,
                                              float* __restrict__ out) {
    __shared__ char  sA[2][BM * LDSS];
    __shared__ char  sB[2][BN * LDSS];
    __shared__ float sP[BM * 20];

    int tid  = threadIdx.x;
    int lane = tid & 31, wave = tid >> 5;
    int wm = wave >> 2, wn = wave & 3;
    int half = lane >> 4, lr = lane & 15;
    int mBase = blockIdx.y * BM;
    int nBase = blockIdx.x * BN;

    for (int i = tid; i < BM * 20; i += 256) sP[i] = params[(size_t)mBase * 20 + i];

    // Per-thread tile chunk coordinates (2 chunks of A + 2 of B per tile)
    int r0 = tid >> 2, c0 = tid & 3;          // rows 0..63
    int r1 = r0 + 64;                         // rows 64..127
    const uint16_t* gA = agg + (size_t)mBase * Dm + c0 * 8;
    const uint16_t* gB = wb  + (size_t)nBase * Dm + c0 * 8;

    v8f acc[4][2];
#pragma unroll
    for (int mi = 0; mi < 4; mi++)
#pragma unroll
        for (int ni = 0; ni < 2; ni++) acc[mi][ni] = (v8f){0, 0, 0, 0, 0, 0, 0, 0};

    uint4 ra0, ra1, rb0, rb1;
    // prologue: tile 0 -> regs -> buf 0
    ra0 = *(const uint4*)(gA + (size_t)r0 * Dm);
    ra1 = *(const uint4*)(gA + (size_t)r1 * Dm);
    rb0 = *(const uint4*)(gB + (size_t)r0 * Dm);
    rb1 = *(const uint4*)(gB + (size_t)r1 * Dm);
    *(uint4*)(sA[0] + r0 * LDSS + c0 * 16) = ra0;
    *(uint4*)(sA[0] + r1 * LDSS + c0 * 16) = ra1;
    *(uint4*)(sB[0] + r0 * LDSS + c0 * 16) = rb0;
    *(uint4*)(sB[0] + r1 * LDSS + c0 * 16) = rb1;

    const int NSTEP = Dm / BK;   // 64
    for (int s = 0; s < NSTEP; s++) {
        int cur = s & 1;
        int kk_next = (s + 1) * BK;
        bool has_next = kk_next < Dm;
        if (has_next) {
            ra0 = *(const uint4*)(gA + (size_t)r0 * Dm + kk_next);
            ra1 = *(const uint4*)(gA + (size_t)r1 * Dm + kk_next);
            rb0 = *(const uint4*)(gB + (size_t)r0 * Dm + kk_next);
            rb1 = *(const uint4*)(gB + (size_t)r1 * Dm + kk_next);
            if (kk_next + BK < Dm) {   // global_prefetch_b8 one more tile ahead
                __builtin_prefetch((const char*)(gA + (size_t)r0 * Dm + kk_next + BK), 0, 1);
                __builtin_prefetch((const char*)(gB + (size_t)r0 * Dm + kk_next + BK), 0, 1);
            }
        }
        __syncthreads();   // buf[cur] ready; everyone done reading buf[1-cur]

        // A fragment: 16x32 bf16 — lane<16 holds K[0..7],K[16..23]; lane>=16 K[8..15],K[24..31]
        Frag a[4];
#pragma unroll
        for (int mi = 0; mi < 4; mi++) {
            const char* p = sA[cur] + (wm * 64 + mi * 16 + lr) * LDSS + half * 16;
            a[mi].u4[0] = *(const uint4*)p;
            a[mi].u4[1] = *(const uint4*)(p + 32);
        }
        // B fragment: 32x16 bf16 — lane<16 holds K[0..7],K[16..23]; lane>=16 K[8..15],K[24..31] of column N
        Frag b[2];
#pragma unroll
        for (int ni = 0; ni < 2; ni++) {
            const char* p = sB[cur] + (wn * 32 + ni * 16 + lr) * LDSS + half * 16;
            b[ni].u4[0] = *(const uint4*)p;
            b[ni].u4[1] = *(const uint4*)(p + 32);
        }
#pragma unroll
        for (int mi = 0; mi < 4; mi++)
#pragma unroll
            for (int ni = 0; ni < 2; ni++)
                acc[mi][ni] = __builtin_amdgcn_wmma_f32_16x16x32_bf16(
                    false, a[mi].v, false, b[ni].v, (short)0, acc[mi][ni], false, false);

        if (has_next) {   // stage tile s+1 into the other buffer
            char* dA = sA[1 - cur];
            char* dB = sB[1 - cur];
            *(uint4*)(dA + r0 * LDSS + c0 * 16) = ra0;
            *(uint4*)(dA + r1 * LDSS + c0 * 16) = ra1;
            *(uint4*)(dB + r0 * LDSS + c0 * 16) = rb0;
            *(uint4*)(dB + r1 * LDSS + c0 * 16) = rb1;
        }
    }

    // Epilogue: out[tok][w][n] = sum_j res[w][j]*x[tok][j][n] + post[w]*(op_out + op_b[n])
#pragma unroll
    for (int ni = 0; ni < 2; ni++) {
        int n = nBase + wn * 32 + ni * 16 + lr;
        float obv = opb[n];
#pragma unroll
        for (int mi = 0; mi < 4; mi++) {
#pragma unroll
            for (int v = 0; v < 8; v++) {
                int ml = wm * 64 + mi * 16 + v + 8 * half;
                int token = mBase + ml;
                float oo = acc[mi][ni][v] + obv;
                const float* pp = sP + ml * 20;
                const float* xp = x + (size_t)token * Wd * Dm + n;
                float x0 = xp[0], x1 = xp[Dm], x2 = xp[2 * Dm], x3 = xp[3 * Dm];
                float* op = out + (size_t)token * Wd * Dm + n;
#pragma unroll
                for (int w = 0; w < 4; w++) {
                    op[w * Dm] = pp[4 + w * 4 + 0] * x0 + pp[4 + w * 4 + 1] * x1 +
                                 pp[4 + w * 4 + 2] * x2 + pp[4 + w * 4 + 3] * x3 +
                                 pp[w] * oo;
                }
            }
        }
    }
}

extern "C" void kernel_launch(void* const* d_in, const int* in_sizes, int n_in,
                              void* d_out, int out_size, void* d_ws, size_t ws_size,
                              hipStream_t stream) {
    const float* x      = (const float*)d_in[0];
    const float* rw     = (const float*)d_in[1];
    const float* rb     = (const float*)d_in[2];
    const float* resb   = (const float*)d_in[3];
    const float* op_w   = (const float*)d_in[4];
    const float* op_b   = (const float*)d_in[5];
    float* out = (float*)d_out;

    char* ws = (char*)d_ws;
    uint16_t* agg    = (uint16_t*)ws;                                   // 32 MB
    uint16_t* wbuf   = (uint16_t*)(ws + (size_t)NTOK * Dm * 2);         // 8 MB
    float*    params = (float*)(ws + (size_t)NTOK * Dm * 2 + (size_t)Dm * Dm * 2); // 640 KB

    k_convert<<<(Dm * Dm) / (256 * 8), 256, 0, stream>>>(op_w, wbuf);
    k_router<<<NTOK / 32, 256, 0, stream>>>(x, rw, rb, resb, agg, params);
    k_gemm<<<dim3(Dm / BN, NTOK / BM), 256, 0, stream>>>(agg, wbuf, op_b, params, x, out);
}
